// Attention_48919677501914
// MI455X (gfx1250) — matrix operand order repaired and checked by reading, not compiled, not run on hardware
//
#include <hip/hip_runtime.h>
#include <hip/hip_bf16.h>

#define S_TXT 512
#define S_IMG 2048
#define S_TOT 2560
#define HEADS 24
#define DH    128
#define QDIM  3072
#define NQKV  9216
#define RANK  32
#define INNER (HEADS * DH)
#define ATTN_SCALE 0.08838834764831845f   // 1/sqrt(128)

typedef __attribute__((ext_vector_type(16))) __bf16 v16bf;
typedef __attribute__((ext_vector_type(8)))  float  v8f;
typedef __attribute__((ext_vector_type(4)))  unsigned int v4u;
typedef __attribute__((ext_vector_type(8)))  int v8i;
typedef __attribute__((ext_vector_type(4)))  int v4i;

#if __has_builtin(__builtin_amdgcn_tensor_load_to_lds)
#define HAVE_TDM 1
#else
#define HAVE_TDM 0
#endif

union Frag16 { v16bf v; unsigned int u[8]; };

__device__ __forceinline__ unsigned short f32_to_bf16(float f) {
    unsigned int u = __builtin_bit_cast(unsigned int, f);
    u += 0x7FFFu + ((u >> 16) & 1u);          // round-to-nearest-even
    return (unsigned short)(u >> 16);
}

// ---------------------------------------------------------------------------
// Generic SVDQ GEMM:  C[M,N] = A[M,K] @ W[K,N]  (+ R[M,32] @ PU[32,N]) (+ bias)
// BM=BN=128, BK=32, 256 threads = 8 waves (2x4), each wave 64x32 = 4x2 WMMA tiles
// fp32 in, bf16 WMMA (v_wmma_f32_16x16x32_bf16), fp32 accumulate/out.
// ---------------------------------------------------------------------------
__global__ __launch_bounds__(256) void gemm_svdq_bf16(
    const float* __restrict__ A, int lda,
    const float* __restrict__ W, int ldw,
    const float* __restrict__ R,
    const float* __restrict__ PU,
    const float* __restrict__ bias,
    float* __restrict__ C, int ldc,
    int M, int N, int K)
{
    __shared__ alignas(16) unsigned short As[128][34];   // [m][k] bf16
    __shared__ alignas(16) unsigned short Bs[128][34];   // [n][k] bf16 (B transposed)

    const int t = threadIdx.x;
    const int w = t >> 5, L = t & 31;
    const int wm = w >> 2, wn = w & 3;
    const int half = L >> 4, lane16 = L & 15;
    const int m0 = blockIdx.y * 128, n0 = blockIdx.x * 128;

    const v8f zero = {0.f,0.f,0.f,0.f,0.f,0.f,0.f,0.f};
    v8f acc[4][2];
#pragma unroll
    for (int i = 0; i < 4; ++i) {
#pragma unroll
        for (int j = 0; j < 2; ++j) acc[i][j] = zero;
    }

    const int nK = K >> 5;
    const int iters = nK + ((R != nullptr) ? 1 : 0);

    for (int kt = 0; kt < iters; ++kt) {
        const bool lr = (kt == nK);                 // low-rank epilogue K-tile
        const float* Asrc = lr ? R  : A;
        const float* Bsrc = lr ? PU : W;
        const int alda = lr ? RANK : lda;
        const int kk   = lr ? 0    : (kt << 5);

        // speculative prefetch of next K-tile (global_prefetch_b8, RT)
        if (kt + 1 < nK) {
            long nk = (long)(kt + 1) << 5;
            __builtin_prefetch(&A[(long)(m0 + (t >> 1)) * lda + nk + ((t & 1) << 4)], 0, 1);
            __builtin_prefetch(&W[(nk + (t >> 3)) * ldw + n0 + ((t & 7) << 4)], 0, 1);
        }

        // stage A tile 128x32 (coalesced along k)
#pragma unroll
        for (int i = 0; i < 16; ++i) {
            int idx = t + (i << 8);
            int r = idx >> 5, c = idx & 31;
            As[r][c] = f32_to_bf16(Asrc[(long)(m0 + r) * alda + kk + c]);
        }
        // stage B tile 32x128 transposed into [n][k] (coalesced along n)
#pragma unroll
        for (int i = 0; i < 16; ++i) {
            int idx = t + (i << 8);
            int kr = idx >> 7, n = idx & 127;
            int gn = n0 + n;
            float vv = (gn < N) ? Bsrc[(long)(kk + kr) * ldw + gn] : 0.f;
            Bs[n][kr] = f32_to_bf16(vv);
        }
        __syncthreads();

        Frag16 af[4], bfr[2];
#pragma unroll
        for (int mi = 0; mi < 4; ++mi) {
            int mr = wm * 64 + mi * 16 + lane16;
#pragma unroll
            for (int v = 0; v < 8; ++v) {
                int k0 = 2 * (v & 3) + 8 * half + ((v >> 2) << 4);
                af[mi].u[v] = *(const unsigned int*)&As[mr][k0];
            }
        }
#pragma unroll
        for (int ni = 0; ni < 2; ++ni) {
            int nc = wn * 32 + ni * 16 + lane16;
#pragma unroll
            for (int v = 0; v < 8; ++v) {
                int k0 = 2 * (v & 3) + 8 * half + ((v >> 2) << 4);
                bfr[ni].u[v] = *(const unsigned int*)&Bs[nc][k0];
            }
        }
#pragma unroll
        for (int mi = 0; mi < 4; ++mi) {
#pragma unroll
            for (int ni = 0; ni < 2; ++ni) {
                acc[mi][ni] = __builtin_amdgcn_wmma_f32_16x16x32_bf16(
                    false, af[mi].v, false, bfr[ni].v,
                    (short)0, acc[mi][ni], false, false);
            }
        }
        __syncthreads();
    }

    // store (C layout: row = r + 8*half, col = lane16)
#pragma unroll
    for (int mi = 0; mi < 4; ++mi) {
#pragma unroll
        for (int ni = 0; ni < 2; ++ni) {
            int gm = m0 + wm * 64 + mi * 16 + 8 * half;
            int gn = n0 + wn * 32 + ni * 16 + lane16;
            if (gn < N) {
                float bv = bias ? bias[gn] : 0.f;
#pragma unroll
                for (int r = 0; r < 8; ++r)
                    C[(long)(gm + r) * ldc + gn] = acc[mi][ni][r] + bv;
            }
        }
    }
}

// ---------------------------------------------------------------------------
// Fused per-head RMS-norm + RoPE + bf16 pack.  Q gets 1/sqrt(DH) folded in.
// One 128-thread block per (token, head).  Output layout: [head][seq][d].
// ---------------------------------------------------------------------------
__global__ __launch_bounds__(128) void normrope_pack(
    const float* __restrict__ qkv_img,
    const float* __restrict__ qkv_txt,
    const float* __restrict__ rope,
    const float* __restrict__ g_q,  const float* __restrict__ g_k,
    const float* __restrict__ g_aq, const float* __restrict__ g_ak,
    unsigned short* __restrict__ qb,
    unsigned short* __restrict__ kb,
    unsigned short* __restrict__ vb)
{
    const int s = blockIdx.x, h = blockIdx.y, d = threadIdx.x;
    const float* src; int row; const float *gq, *gk;
    if (s < S_TXT) { src = qkv_txt; row = s;         gq = g_aq; gk = g_ak; }
    else           { src = qkv_img; row = s - S_TXT; gq = g_q;  gk = g_k;  }

    long base = (long)row * NQKV + h * DH + d;
    float qv = src[base];
    float kv = src[base + QDIM];
    float vv = src[base + 2 * QDIM];

    __shared__ float sq[128], sk[128], qn[128], kn[128];
    sq[d] = qv * qv;
    sk[d] = kv * kv;
    __syncthreads();
    for (int s2 = 64; s2 > 0; s2 >>= 1) {
        if (d < s2) { sq[d] += sq[d + s2]; sk[d] += sk[d + s2]; }
        __syncthreads();
    }
    float rq = rsqrtf(sq[0] * (1.f / DH) + 1e-5f);
    float rk = rsqrtf(sk[0] * (1.f / DH) + 1e-5f);
    qn[d] = qv * rq * gq[d];
    kn[d] = kv * rk * gk[d];
    __syncthreads();

    int p = d >> 1;
    float cs = rope[((long)s * (DH / 2) + p) * 2 + 0];
    float sn = rope[((long)s * (DH / 2) + p) * 2 + 1];
    float qo, ko;
    if ((d & 1) == 0) { qo = qn[d] * cs - qn[d + 1] * sn; ko = kn[d] * cs - kn[d + 1] * sn; }
    else              { qo = qn[d - 1] * sn + qn[d] * cs; ko = kn[d - 1] * sn + kn[d] * cs; }

    long o = ((long)h * S_TOT + s) * DH + d;
    qb[o] = f32_to_bf16(qo * ATTN_SCALE);
    kb[o] = f32_to_bf16(ko);
    vb[o] = f32_to_bf16(vv);
}

// ---------------------------------------------------------------------------
// Flash attention.  Block = 8 waves, 128 query rows (16 per wave), KV tile 64.
// K tile staged by the Tensor Data Mover (TDM) with LDS row padding applied by
// the D# pad feature; P*V and QK^T via v_wmma_f32_16x16x32_bf16.
// ---------------------------------------------------------------------------
__global__ __launch_bounds__(256) void flash_attn(
    const unsigned short* __restrict__ qb,
    const unsigned short* __restrict__ kb,
    const unsigned short* __restrict__ vb,
    float* __restrict__ attn)
{
    __shared__ alignas(16) unsigned short Ks[64][136];     // [kv][d] (pad = 8 bf16/row)
    __shared__ alignas(16) unsigned short Vt[128][72];     // [d][kv] (transposed)
    __shared__ alignas(16) unsigned short Ps[8][16][72];   // per-wave P tile [m][kv]

    const int h   = blockIdx.x;
    const int qb0 = blockIdx.y * 128;
    const int t = threadIdx.x;
    const int w = t >> 5, L = t & 31;
    const int half = L >> 4, lane16 = L & 15;

    // Load Q fragments once (A-layout, pre-scaled bf16)
    Frag16 qa[4];
    {
        int sq_ = qb0 + w * 16 + lane16;
        const unsigned short* qrow = qb + ((long)h * S_TOT + sq_) * DH;
#pragma unroll
        for (int f = 0; f < 4; ++f) {
#pragma unroll
            for (int v = 0; v < 8; ++v) {
                int k0 = 32 * f + 2 * (v & 3) + 8 * half + ((v >> 2) << 4);
                qa[f].u[v] = *(const unsigned int*)&qrow[k0];
            }
        }
    }

    const v8f zero = {0.f,0.f,0.f,0.f,0.f,0.f,0.f,0.f};
    v8f o[8];
#pragma unroll
    for (int i = 0; i < 8; ++i) o[i] = zero;
    float m_i[8], l_i[8];
#pragma unroll
    for (int r = 0; r < 8; ++r) { m_i[r] = -1e30f; l_i[r] = 0.f; }

    for (int kv0 = 0; kv0 < S_TOT; kv0 += 64) {
#if HAVE_TDM
        // --- TDM: DMA the 64x128 bf16 K tile into padded LDS rows (wave 0) ---
        if (t < 32) {
            unsigned long long ga =
                (unsigned long long)(const void*)(kb + ((long)h * S_TOT + kv0) * DH);
            unsigned int lds = (unsigned int)(unsigned long long)(const void*)&Ks[0][0];
            // D# group0: count=1 | lds_addr | global_addr[56:0] | type=2
            v4u g0 = { 1u, lds,
                       (unsigned int)ga,
                       (unsigned int)((ga >> 32) & 0x01FFFFFFull) | 0x80000000u };
            // D# group1: data_size=2B, pad every 64 DWORDs by 4 DWORDs,
            //            tensor_dim0=128, tensor_dim1=64, tile 128x64, stride0=128
            unsigned long long q0 = (1ull << 16)          // data_size = 2 bytes
                                  | (1ull << 20)          // pad_enable
                                  | (5ull << 22)          // pad_interval: 64 DWORDs
                                  | (3ull << 25)          // pad_amount: 4 DWORDs
                                  | (128ull << 48);       // tensor_dim0[15:0]
            unsigned long long q1 = 0ull                  // tensor_dim0[31:16]
                                  | (64ull  << 16)        // tensor_dim1
                                  | (128ull << 48);       // tile_dim0
            unsigned long long q2 = 64ull                 // tile_dim1
                                  | (128ull << 32);       // tensor_dim0_stride[31:0]
            unsigned long long q3 = 0ull;                 // stride0 hi / stride1
            union { unsigned long long q[4]; v8i v; } g1u;
            g1u.q[0] = q0; g1u.q[1] = q1; g1u.q[2] = q2; g1u.q[3] = q3;
            v4i gz = {0, 0, 0, 0};
#if __has_include(<hip/amd_detail/amd_gfx1250_TDM.h>)
            v8i gz8 = {0, 0, 0, 0, 0, 0, 0, 0};
            __builtin_amdgcn_tensor_load_to_lds(g0, g1u.v, gz, gz, gz8, 0);
#else
            __builtin_amdgcn_tensor_load_to_lds(g0, g1u.v, gz, gz, 0);
#endif
        }
#else
        // fallback: manual K staging [kv][d]
#pragma unroll
        for (int i = 0; i < 16; ++i) {
            int idx = t + (i << 8);
            int cp = idx & 63, r = idx >> 6;
            unsigned int u = *(const unsigned int*)&kb[((long)h * S_TOT + kv0 + r) * DH + 2 * cp];
            *(unsigned int*)&Ks[r][2 * cp] = u;
        }
#endif
        // stage V transposed [d][kv] (TDM cannot transpose)
#pragma unroll
        for (int i = 0; i < 16; ++i) {
            int idx = t + (i << 8);
            int cp = idx & 63, r = idx >> 6;
            unsigned int u = *(const unsigned int*)&vb[((long)h * S_TOT + kv0 + r) * DH + 2 * cp];
            Vt[2 * cp][r]     = (unsigned short)(u & 0xFFFFu);
            Vt[2 * cp + 1][r] = (unsigned short)(u >> 16);
        }
#if HAVE_TDM
        if (t < 32) __builtin_amdgcn_s_wait_tensorcnt(0);   // TENSORcnt is per-wave
#endif
        __syncthreads();

        // S = Q @ K^T : per wave 16x64 = 4 n-tiles, K-dim 128 = 4 steps
        v8f sc[4];
#pragma unroll
        for (int nt = 0; nt < 4; ++nt) sc[nt] = zero;
#pragma unroll
        for (int ks = 0; ks < 4; ++ks) {
            Frag16 bk[4];
#pragma unroll
            for (int nt = 0; nt < 4; ++nt) {
                int n = nt * 16 + lane16;
#pragma unroll
                for (int v = 0; v < 8; ++v) {
                    int k0 = 32 * ks + 2 * (v & 3) + 8 * half + ((v >> 2) << 4);
                    bk[nt].u[v] = *(const unsigned int*)&Ks[n][k0];
                }
            }
#pragma unroll
            for (int nt = 0; nt < 4; ++nt) {
                sc[nt] = __builtin_amdgcn_wmma_f32_16x16x32_bf16(
                    false, qa[ks].v, false, bk[nt].v, (short)0, sc[nt], false, false);
            }
        }

        // online softmax update (row r lives in vgpr r, half-group of 16 lanes)
#pragma unroll
        for (int r = 0; r < 8; ++r) {
            float rmax = sc[0][r];
#pragma unroll
            for (int nt = 1; nt < 4; ++nt) rmax = fmaxf(rmax, sc[nt][r]);
#pragma unroll
            for (int msk = 1; msk <= 8; msk <<= 1)
                rmax = fmaxf(rmax, __shfl_xor(rmax, msk, 32));
            float m_new = fmaxf(m_i[r], rmax);
            float alpha = __expf(m_i[r] - m_new);
            float psum = 0.f;
#pragma unroll
            for (int nt = 0; nt < 4; ++nt) {
                float p = __expf(sc[nt][r] - m_new);
                sc[nt][r] = p;
                psum += p;
            }
#pragma unroll
            for (int msk = 1; msk <= 8; msk <<= 1)
                psum += __shfl_xor(psum, msk, 32);
            l_i[r] = l_i[r] * alpha + psum;
            m_i[r] = m_new;
#pragma unroll
            for (int nt2 = 0; nt2 < 8; ++nt2) o[nt2][r] *= alpha;
        }

        // P (C-layout) -> per-wave LDS (A-layout source)
#pragma unroll
        for (int nt = 0; nt < 4; ++nt) {
#pragma unroll
            for (int r = 0; r < 8; ++r)
                Ps[w][r + 8 * half][nt * 16 + lane16] = f32_to_bf16(sc[nt][r]);
        }

        // O += P @ V : K-dim 64 = 2 steps, 8 d-tiles
#pragma unroll
        for (int ks2 = 0; ks2 < 2; ++ks2) {
            Frag16 pa;
#pragma unroll
            for (int v = 0; v < 8; ++v) {
                int k0 = 32 * ks2 + 2 * (v & 3) + 8 * half + ((v >> 2) << 4);
                pa.u[v] = *(const unsigned int*)&Ps[w][lane16][k0];
            }
#pragma unroll
            for (int nt2 = 0; nt2 < 8; ++nt2) {
                Frag16 bv2;
                int n = nt2 * 16 + lane16;
#pragma unroll
                for (int v = 0; v < 8; ++v) {
                    int k0 = 32 * ks2 + 2 * (v & 3) + 8 * half + ((v >> 2) << 4);
                    bv2.u[v] = *(const unsigned int*)&Vt[n][k0];
                }
                o[nt2] = __builtin_amdgcn_wmma_f32_16x16x32_bf16(
                    false, pa.v, false, bv2.v, (short)0, o[nt2], false, false);
            }
        }
        __syncthreads();
    }

    // epilogue: O / l, store attn[s][h*DH + d] fp32
#pragma unroll
    for (int r = 0; r < 8; ++r) {
        float inv = 1.f / l_i[r];
        int srow = qb0 + w * 16 + 8 * half + r;
#pragma unroll
        for (int nt2 = 0; nt2 < 8; ++nt2) {
            int dcol = nt2 * 16 + lane16;
            attn[(long)srow * INNER + h * DH + dcol] = o[nt2][r] * inv;
        }
    }
}

// ---------------------------------------------------------------------------
extern "C" void kernel_launch(void* const* d_in, const int* in_sizes, int n_in,
                              void* d_out, int out_size, void* d_ws, size_t ws_size,
                              hipStream_t stream)
{
    (void)in_sizes; (void)n_in; (void)out_size; (void)ws_size;
    const float* hs      = (const float*)d_in[0];
    const float* ehs     = (const float*)d_in[1];
    const float* rope    = (const float*)d_in[2];
    const float* w_qkv   = (const float*)d_in[3];
    const float* pd_qkv  = (const float*)d_in[4];
    const float* pu_qkv  = (const float*)d_in[5];
    const float* w_aqkv  = (const float*)d_in[6];
    const float* pd_aqkv = (const float*)d_in[7];
    const float* pu_aqkv = (const float*)d_in[8];
    const float* w_out   = (const float*)d_in[9];
    const float* pd_out  = (const float*)d_in[10];
    const float* pu_out  = (const float*)d_in[11];
    const float* b_out   = (const float*)d_in[12];
    const float* w_aout  = (const float*)d_in[13];
    const float* pd_aout = (const float*)d_in[14];
    const float* pu_aout = (const float*)d_in[15];
    const float* b_aout  = (const float*)d_in[16];
    const float* g_q     = (const float*)d_in[17];
    const float* g_k     = (const float*)d_in[18];
    const float* g_aq    = (const float*)d_in[19];
    const float* g_ak    = (const float*)d_in[20];

    char* ws = (char*)d_ws;
    size_t off = 0;
    auto alloc = [&](size_t bytes) -> void* {
        void* p = ws + off;
        off += (bytes + 255) & ~(size_t)255;
        return p;
    };
    float* qkv_img = (float*)alloc((size_t)S_IMG * NQKV * sizeof(float));
    float* qkv_txt = (float*)alloc((size_t)S_TXT * NQKV * sizeof(float));
    float* rankb   = (float*)alloc((size_t)S_IMG * RANK * sizeof(float));
    unsigned short* qbuf = (unsigned short*)alloc((size_t)HEADS * S_TOT * DH * 2);
    unsigned short* kbuf = (unsigned short*)alloc((size_t)HEADS * S_TOT * DH * 2);
    unsigned short* vbuf = (unsigned short*)alloc((size_t)HEADS * S_TOT * DH * 2);
    float* attn = (float*)alloc((size_t)S_TOT * INNER * sizeof(float));

    const dim3 blk(256);

    // ---- QKV projections (rank GEMM, then main GEMM with fused low-rank) ----
    gemm_svdq_bf16<<<dim3(1, S_IMG / 128), blk, 0, stream>>>(
        hs, QDIM, pd_qkv, RANK, nullptr, nullptr, nullptr,
        rankb, RANK, S_IMG, RANK, QDIM);
    gemm_svdq_bf16<<<dim3(NQKV / 128, S_IMG / 128), blk, 0, stream>>>(
        hs, QDIM, w_qkv, NQKV, rankb, pu_qkv, nullptr,
        qkv_img, NQKV, S_IMG, NQKV, QDIM);

    gemm_svdq_bf16<<<dim3(1, S_TXT / 128), blk, 0, stream>>>(
        ehs, QDIM, pd_aqkv, RANK, nullptr, nullptr, nullptr,
        rankb, RANK, S_TXT, RANK, QDIM);
    gemm_svdq_bf16<<<dim3(NQKV / 128, S_TXT / 128), blk, 0, stream>>>(
        ehs, QDIM, w_aqkv, NQKV, rankb, pu_aqkv, nullptr,
        qkv_txt, NQKV, S_TXT, NQKV, QDIM);

    // ---- RMS-norm + RoPE + pack to bf16 [head][seq][d] ----
    normrope_pack<<<dim3(S_TOT, HEADS), dim3(128), 0, stream>>>(
        qkv_img, qkv_txt, rope, g_q, g_k, g_aq, g_ak, qbuf, kbuf, vbuf);

    // ---- Flash attention ----
    flash_attn<<<dim3(HEADS, S_TOT / 128), blk, 0, stream>>>(qbuf, kbuf, vbuf, attn);

    // ---- Output projections ----
    const float* attn_img = attn + (size_t)S_TXT * INNER;
    float* out_img = (float*)d_out;
    float* out_txt = (float*)d_out + (size_t)S_IMG * QDIM;

    gemm_svdq_bf16<<<dim3(1, S_IMG / 128), blk, 0, stream>>>(
        attn_img, INNER, pd_out, RANK, nullptr, nullptr, nullptr,
        rankb, RANK, S_IMG, RANK, INNER);
    gemm_svdq_bf16<<<dim3(QDIM / 128, S_IMG / 128), blk, 0, stream>>>(
        attn_img, INNER, w_out, QDIM, rankb, pu_out, b_out,
        out_img, QDIM, S_IMG, QDIM, INNER);

    gemm_svdq_bf16<<<dim3(1, S_TXT / 128), blk, 0, stream>>>(
        attn, INNER, pd_aout, RANK, nullptr, nullptr, nullptr,
        rankb, RANK, S_TXT, RANK, INNER);
    gemm_svdq_bf16<<<dim3(QDIM / 128, S_TXT / 128), blk, 0, stream>>>(
        attn, INNER, w_aout, QDIM, rankb, pu_aout, b_aout,
        out_txt, QDIM, S_TXT, QDIM, INNER);
}
